// EvoformerBlockCore_89850715833163
// MI455X (gfx1250) — matrix operand order repaired and checked by reading, not compiled, not run on hardware
//
#include <hip/hip_runtime.h>
#include <cstddef>
#include <cstdint>

// ---------------------------------------------------------------------------
// Evoformer block core for MI455X (gfx1250, wave32, WMMA).
// All matmul-shaped ops lower to v_wmma_f32_16x16x32_bf16 (bf16 in, f32 acc).
// Fragments are contiguous b128 LDS transactions; main GEMM is double-buffered.
// ---------------------------------------------------------------------------

typedef __attribute__((ext_vector_type(16))) __bf16 v16bf;
typedef __attribute__((ext_vector_type(8)))  __bf16 v8bf;
typedef __attribute__((ext_vector_type(8)))  float  v8f;
typedef __attribute__((ext_vector_type(4)))  float  v4f;

#define WMMA_BF16(A_, B_, C_) \
  __builtin_amdgcn_wmma_f32_16x16x32_bf16(false, (A_), false, (B_), (short)0, (C_), false, false)

#define N_RES_C 256
#define N_SEQ_C 128
#define CM_C    256
#define CZ_C    128
#define ATT_INF 1e9f
#define OPM_EPS 1e-3f

// Convert 8 consecutive floats -> v8bf (lowers to packed cvt).
__device__ inline v8bf cvt8(const float* __restrict__ p) {
  v4f a = *(const v4f*)p;
  v4f b = *(const v4f*)(p + 4);
  v8bf r;
  #pragma unroll
  for (int i = 0; i < 4; ++i) { r[i] = (__bf16)a[i]; r[i + 4] = (__bf16)b[i]; }
  return r;
}

// A-fragment for 16x16x32 bf16 WMMA: lane holds k = 2*hs..2*hs+15
// of its row. Two contiguous 16B LDS loads.
__device__ inline v16bf frag_from_row(const __bf16* __restrict__ row, int hs) {
  v8bf lo = *(const v8bf*)(row + 2 * hs);
  v8bf hi = *(const v8bf*)(row + 2 * hs + 8);
  return __builtin_shufflevector(lo, hi, 0, 1, 2, 3, 4, 5, 6, 7,
                                 8, 9, 10, 11, 12, 13, 14, 15);
}

// B-fragment: lane holds k = koff..koff+15 of its column: one 32B LDS load.
__device__ inline v16bf frag_contig(const __bf16* __restrict__ p) {
  return *(const v16bf*)p;
}

// ======================= small utility kernels =============================

// Pad-convert weight fp32[K,N] -> bf16[K,Npad] (zero-fill cols >= N).
__global__ void conv_pad_kernel(const float* __restrict__ s, __bf16* __restrict__ d,
                                int K, int N, int Npad) {
  int idx = blockIdx.x * 256 + threadIdx.x;
  if (idx >= K * Npad) return;
  int k = idx / Npad, c = idx % Npad;
  d[idx] = (c < N) ? (__bf16)s[(size_t)k * N + c] : (__bf16)0.f;
}

// y[i] *= (b? b[i] : 1) * (tok? tok[i/C] : 1)
__global__ void ew_mul3_kernel(float* __restrict__ y, const float* __restrict__ b,
                               const float* __restrict__ tok, size_t total, int C) {
  size_t i = (size_t)blockIdx.x * blockDim.x + threadIdx.x;
  if (i >= total) return;
  float v = y[i];
  if (b)   v *= b[i];
  if (tok) v *= tok[i / C];
  y[i] = v;
}

// z[i] += x[i] * (tok? tok[i/C] : 1)
__global__ void ew_resadd_kernel(float* __restrict__ z, const float* __restrict__ x,
                                 const float* __restrict__ tok, size_t total, int C) {
  size_t i = (size_t)blockIdx.x * blockDim.x + threadIdx.x;
  if (i >= total) return;
  float v = x[i];
  if (tok) v *= tok[i / C];
  z[i] += v;
}

__global__ void ew_resmul_kernel(float* __restrict__ z, const float* __restrict__ a,
                                 const float* __restrict__ b, size_t total) {
  size_t i = (size_t)blockIdx.x * blockDim.x + threadIdx.x;
  if (i < total) z[i] += a[i] * b[i];
}

__global__ void ew_opmfinal_kernel(float* __restrict__ z, const float* __restrict__ x,
                                   const float* __restrict__ norm, size_t total, int C) {
  size_t i = (size_t)blockIdx.x * blockDim.x + threadIdx.x;
  if (i < total) z[i] += x[i] / (norm[i / C] + OPM_EPS);
}

__global__ void transpose_nn_kernel(const float* __restrict__ x, float* __restrict__ y,
                                    int n, int C) {
  size_t idx = (size_t)blockIdx.x * blockDim.x + threadIdx.x;
  size_t total = (size_t)n * n * C;
  if (idx >= total) return;
  int c = (int)(idx % C);
  size_t ij = idx / C;
  int j = (int)(ij % n);
  int i = (int)(ij / n);
  y[idx] = x[((size_t)j * n + i) * C + c];
}

__global__ void ew_addT_kernel(float* __restrict__ z, const float* __restrict__ x,
                               int n, int C) {
  size_t idx = (size_t)blockIdx.x * blockDim.x + threadIdx.x;
  size_t total = (size_t)n * n * C;
  if (idx >= total) return;
  int c = (int)(idx % C);
  size_t ij = idx / C;
  int j = (int)(ij % n);
  int i = (int)(ij / n);
  z[idx] += x[((size_t)j * n + i) * C + c];
}

__global__ void norm_outer_kernel(const float* __restrict__ mm, float* __restrict__ norm,
                                  int S, int n) {
  int idx = blockIdx.x * blockDim.x + threadIdx.x;
  if (idx >= n * n) return;
  int i = idx / n, j = idx % n;
  float s = 0.f;
  for (int t = 0; t < S; ++t) s += mm[(size_t)t * n + i] * mm[(size_t)t * n + j];
  norm[idx] = s;
}

// LayerNorm over rows of width C (C <= 256); one block (256 thr) per row.
__global__ __launch_bounds__(256) void ln_kernel(const float* __restrict__ x,
                                                 const float* __restrict__ g,
                                                 const float* __restrict__ b,
                                                 float* __restrict__ y, int C) {
  __shared__ float red[256];
  size_t row = blockIdx.x;
  int t = threadIdx.x;
  float v = (t < C) ? x[row * C + t] : 0.f;
  red[t] = v; __syncthreads();
  for (int s = 128; s > 0; s >>= 1) { if (t < s) red[t] += red[t + s]; __syncthreads(); }
  float mu = red[0] / C;
  __syncthreads();
  float d = (t < C) ? (v - mu) : 0.f;
  red[t] = d * d; __syncthreads();
  for (int s = 128; s > 0; s >>= 1) { if (t < s) red[t] += red[t + s]; __syncthreads(); }
  float var = red[0] / C;
  float rstd = rsqrtf(var + 1e-5f);
  if (t < C) y[row * C + t] = d * rstd * g[t] + b[t];
}

// ======================= generic WMMA GEMM =================================
// Y[M,N] = act( A[M,K] @ W[K,Npad] + bias[N] ), W bf16 zero-padded to Npad%64==0.
// M%128==0, K%32==0.  Block: 256 thr (8 waves), 128x64 tile; each wave owns
// 16 rows x 64 cols (4 WMMA per k-step).  Double-buffered LDS, 1 barrier/step.
template <int A_BF16>
__global__ __launch_bounds__(256) void gemm_bf16_kernel(
    const float* __restrict__ A, const __bf16* __restrict__ Abf,
    const __bf16* __restrict__ W, const float* __restrict__ bias,
    float* __restrict__ Y, int M, int N, int Npad, int K, int act) {
  __shared__ __bf16 As[2][128][32];  // [buf][row][k]
  __shared__ __bf16 Wt[2][64][32];   // [buf][col][k] (transposed W tile)
  const int tid = threadIdx.x;
  const int wave = tid >> 5, lane = tid & 31;
  const int row0 = blockIdx.y * 128 + wave * 16;
  const int col0 = blockIdx.x * 64;
  const int lrow = lane & 15;
  const int hs = (lane >= 16) ? 8 : 0;
  const int koff = (lane >= 16) ? 16 : 0;
  // staging coordinates: A = 128 rows x 4 chunks (2 tasks/thread);
  //                      W = 32 k x 8 chunks (1 task/thread).
  const int ar0 = tid >> 2, akq0 = (tid & 3) * 8;
  const int wkk = tid >> 3, wcq = (tid & 7) * 8;
  const size_t arow0 = (size_t)(blockIdx.y * 128 + ar0) * K;
  const size_t arow1 = arow0 + (size_t)64 * K;  // second task: row + 64

  v8bf areg0, areg1, wreg;
  // ---- prologue: tile 0 ----
  if (A_BF16) {
    areg0 = *(const v8bf*)&Abf[arow0 + akq0];
    areg1 = *(const v8bf*)&Abf[arow1 + akq0];
  } else {
    areg0 = cvt8(&A[arow0 + akq0]);
    areg1 = cvt8(&A[arow1 + akq0]);
  }
  wreg = *(const v8bf*)&W[(size_t)wkk * Npad + col0 + wcq];
  *(v8bf*)&As[0][ar0][akq0] = areg0;
  *(v8bf*)&As[0][64 + ar0][akq0] = areg1;
  #pragma unroll
  for (int u = 0; u < 8; ++u) Wt[0][wcq + u][wkk] = wreg[u];

  v8f acc0 = {}, acc1 = {}, acc2 = {}, acc3 = {};
  int buf = 0;
  for (int k0 = 0; k0 < K; k0 += 32) {
    __syncthreads();
    const bool more = (k0 + 32) < K;
    if (more) {  // load next tile into registers (overlaps with WMMA below)
      if (A_BF16) {
        areg0 = *(const v8bf*)&Abf[arow0 + k0 + 32 + akq0];
        areg1 = *(const v8bf*)&Abf[arow1 + k0 + 32 + akq0];
      } else {
        areg0 = cvt8(&A[arow0 + k0 + 32 + akq0]);
        areg1 = cvt8(&A[arow1 + k0 + 32 + akq0]);
        if (k0 + 64 < K) __builtin_prefetch(&A[arow0 + k0 + 64 + akq0], 0, 1);
      }
      wreg = *(const v8bf*)&W[(size_t)(k0 + 32 + wkk) * Npad + col0 + wcq];
    }
    // compute from current buffer
    v16bf af = frag_from_row(&As[buf][wave * 16 + lrow][0], hs);
    acc0 = WMMA_BF16(af, frag_contig(&Wt[buf][lrow][koff]),      acc0);
    acc1 = WMMA_BF16(af, frag_contig(&Wt[buf][16 + lrow][koff]), acc1);
    acc2 = WMMA_BF16(af, frag_contig(&Wt[buf][32 + lrow][koff]), acc2);
    acc3 = WMMA_BF16(af, frag_contig(&Wt[buf][48 + lrow][koff]), acc3);
    if (more) {  // store next tile into the other buffer
      *(v8bf*)&As[buf ^ 1][ar0][akq0] = areg0;
      *(v8bf*)&As[buf ^ 1][64 + ar0][akq0] = areg1;
      #pragma unroll
      for (int u = 0; u < 8; ++u) Wt[buf ^ 1][wcq + u][wkk] = wreg[u];
    }
    buf ^= 1;
  }
  #pragma unroll
  for (int t = 0; t < 4; ++t) {
    const v8f* accp = (t == 0) ? &acc0 : (t == 1) ? &acc1 : (t == 2) ? &acc2 : &acc3;
    #pragma unroll
    for (int r = 0; r < 8; ++r) {
      int row = row0 + r + hs;
      int col = col0 + t * 16 + lrow;
      if (col >= N) continue;
      float v = (*accp)[r] + bias[col];
      if (act == 1) v = fmaxf(v, 0.f);
      else if (act == 2) v = 1.f / (1.f + __expf(-v));
      Y[(size_t)row * N + col] = v;
    }
  }
}

// ======================= OPM outer product =================================
// o[i,j,c*32+d] (bf16) = sum_s a[s,i,c]*b[s,j,d].  Block per (i,j), 4 waves.
__global__ __launch_bounds__(128) void opm_outer_kernel(
    const float* __restrict__ a, const float* __restrict__ b,
    __bf16* __restrict__ o, int S, int n) {
  __shared__ __bf16 Asx[32][128];  // [c][s]
  __shared__ __bf16 Bsx[32][128];  // [d][s]
  const int tid = threadIdx.x, wave = tid >> 5, lane = tid & 31;
  const int i = blockIdx.x, j = blockIdx.y;
  for (int e = 0; e < 4; ++e) {
    int task = e * 128 + tid;            // 128 s x 4 chunks
    int ss = task >> 2, cq = (task & 3) * 8;
    v8bf av = cvt8(&a[((size_t)ss * n + i) * 32 + cq]);
    v8bf bv = cvt8(&b[((size_t)ss * n + j) * 32 + cq]);
    #pragma unroll
    for (int u = 0; u < 8; ++u) { Asx[cq + u][ss] = av[u]; Bsx[cq + u][ss] = bv[u]; }
  }
  __syncthreads();
  const int ct = wave >> 1, dt = wave & 1;
  const int lrow = lane & 15;
  const int hs = (lane >= 16) ? 8 : 0;
  const int koff = (lane >= 16) ? 16 : 0;
  v8f acc = {};
  for (int k0 = 0; k0 < S; k0 += 32) {
    v16bf af = frag_from_row(&Asx[ct * 16 + lrow][k0], hs);
    v16bf bf = frag_contig(&Bsx[dt * 16 + lrow][k0 + koff]);
    acc = WMMA_BF16(af, bf, acc);
  }
  #pragma unroll
  for (int r = 0; r < 8; ++r) {
    int c = ct * 16 + r + hs;
    int d = dt * 16 + lrow;
    o[((size_t)i * n + j) * 1024 + c * 32 + d] = (__bf16)acc[r];
  }
}

// ======================= triangle-mul einsum ===============================
// outgoing: x[i,j,c] = sum_k a[i,k,c]*b[j,k,c];  incoming: a[k,i,c]*b[k,j,c]
// Block: 256 thr (8 waves), 16x16 (i,j) tile x 8 channels (one per wave).
__global__ __launch_bounds__(256) void trimul_einsum_kernel(
    const float* __restrict__ a, const float* __restrict__ b,
    float* __restrict__ x, int n, int C, int outgoing) {
  __shared__ __bf16 As[8][16][32];  // [c][i-row][k]
  __shared__ __bf16 Bs[8][16][32];  // [c][j-col][k]
  const int tid = threadIdx.x, wave = tid >> 5, lane = tid & 31;
  const int i0 = blockIdx.x * 16, j0 = blockIdx.y * 16, c0 = blockIdx.z * 8;
  const int lrow = lane & 15;
  const int hs = (lane >= 16) ? 8 : 0;
  const int koff = (lane >= 16) ? 16 : 0;
  v8f acc = {};
  for (int k0 = 0; k0 < n; k0 += 32) {
    for (int e = 0; e < 2; ++e) {
      int task = e * 256 + tid;          // 16 rows x 32 k
      int kk = task & 31, rr = (task >> 5) & 15;
      size_t ai = outgoing ? ((size_t)(i0 + rr) * n + (k0 + kk)) * C + c0
                           : ((size_t)(k0 + kk) * n + (i0 + rr)) * C + c0;
      size_t bi = outgoing ? ((size_t)(j0 + rr) * n + (k0 + kk)) * C + c0
                           : ((size_t)(k0 + kk) * n + (j0 + rr)) * C + c0;
      v8bf av = cvt8(&a[ai]);
      v8bf bv = cvt8(&b[bi]);
      #pragma unroll
      for (int u = 0; u < 8; ++u) { As[u][rr][kk] = av[u]; Bs[u][rr][kk] = bv[u]; }
    }
    __syncthreads();
    v16bf af = frag_from_row(&As[wave][lrow][0], hs);
    v16bf bf = frag_contig(&Bs[wave][lrow][koff]);
    acc = WMMA_BF16(af, bf, acc);
    __syncthreads();
  }
  #pragma unroll
  for (int r = 0; r < 8; ++r) {
    int row = r + hs, col = lrow;
    x[((size_t)(i0 + row) * n + (j0 + col)) * C + c0 + wave] = acc[r];
  }
}

// ======================= fused triangle attention ==========================
// Per block: (i, h, 4 j-tiles); per wave: 16 rows of j; full softmax over p.
__global__ __launch_bounds__(128) void tri_att_kernel(
    const float* __restrict__ q, const float* __restrict__ k,
    const float* __restrict__ v, const float* __restrict__ tb,
    const float* __restrict__ mask, float* __restrict__ o, int n) {
  __shared__ __bf16 Ks[256][32];   // [p][c]
  __shared__ __bf16 Vsx[32][256];  // [c][p]  (transposed for B-fragments)
  __shared__ __bf16 At[4][16][32]; // per-wave attn chunk [j-row][p]
  const int tid = threadIdx.x, wave = tid >> 5, lane = tid & 31;
  const int i = blockIdx.z, h = blockIdx.y;
  const int j0 = (blockIdx.x * 4 + wave) * 16;
  const size_t rowbase = (size_t)i * n * 128 + h * 32;
  for (int e = 0; e < 8; ++e) {
    int task = e * 128 + tid;            // 256 p x 4 chunks
    int pp = task >> 2, cq = (task & 3) * 8;
    *(v8bf*)&Ks[pp][cq] = cvt8(&k[rowbase + (size_t)pp * 128 + cq]);
    v8bf vv = cvt8(&v[rowbase + (size_t)pp * 128 + cq]);
    #pragma unroll
    for (int u = 0; u < 8; ++u) Vsx[cq + u][pp] = vv[u];
  }
  __syncthreads();
  const int lrow = lane & 15;
  const int hs = (lane >= 16) ? 8 : 0;
  const int koff = (lane >= 16) ? 16 : 0;
  // Q fragment straight from global (two 8-float chunks).
  v16bf qf;
  {
    const float* qr = &q[rowbase + (size_t)(j0 + lrow) * 128];
    v8bf lo = cvt8(qr + 2 * hs);
    v8bf hi = cvt8(qr + 2 * hs + 8);
    qf = __builtin_shufflevector(lo, hi, 0, 1, 2, 3, 4, 5, 6, 7,
                                 8, 9, 10, 11, 12, 13, 14, 15);
  }
  v8f sc[16];
  #pragma unroll 4
  for (int pt = 0; pt < 16; ++pt) {
    v16bf kf = frag_contig(&Ks[pt * 16 + lrow][koff]);
    v8f z0 = {};
    sc[pt] = WMMA_BF16(qf, kf, z0);
  }
  const float scale = 0.17677669529663687f;  // 1/sqrt(32)
  float rmax[8], rsum[8];
  #pragma unroll
  for (int r = 0; r < 8; ++r) rmax[r] = -3.0e38f;
  #pragma unroll
  for (int pt = 0; pt < 16; ++pt) {
    int p = pt * 16 + lrow;
    float mb = (mask[(size_t)i * n + p] - 1.0f) * ATT_INF;
    #pragma unroll
    for (int r = 0; r < 8; ++r) {
      int jrow = j0 + r + hs;
      float val = sc[pt][r] * scale + mb + tb[((size_t)jrow * n + p) * 4 + h];
      sc[pt][r] = val;
      rmax[r] = fmaxf(rmax[r], val);
    }
  }
  #pragma unroll
  for (int r = 0; r < 8; ++r) {
    float m = rmax[r];
    m = fmaxf(m, __shfl_xor(m, 1, 32));
    m = fmaxf(m, __shfl_xor(m, 2, 32));
    m = fmaxf(m, __shfl_xor(m, 4, 32));
    m = fmaxf(m, __shfl_xor(m, 8, 32));
    rmax[r] = m; rsum[r] = 0.f;
  }
  #pragma unroll
  for (int pt = 0; pt < 16; ++pt)
    #pragma unroll
    for (int r = 0; r < 8; ++r) {
      float e = __expf(sc[pt][r] - rmax[r]);
      sc[pt][r] = e;
      rsum[r] += e;
    }
  #pragma unroll
  for (int r = 0; r < 8; ++r) {
    float s = rsum[r];
    s += __shfl_xor(s, 1, 32);
    s += __shfl_xor(s, 2, 32);
    s += __shfl_xor(s, 4, 32);
    s += __shfl_xor(s, 8, 32);
    rsum[r] = 1.0f / s;
  }
  v8f oacc0 = {}, oacc1 = {};
  for (int ks = 0; ks < 8; ++ks) {
    #pragma unroll
    for (int half = 0; half < 2; ++half) {
      int pt = ks * 2 + half;
      #pragma unroll
      for (int r = 0; r < 8; ++r)
        At[wave][r + hs][half * 16 + lrow] = (__bf16)(sc[pt][r] * rsum[r]);
    }
    __syncthreads();
    v16bf af = frag_from_row(&At[wave][lrow][0], hs);
    v16bf bf0 = frag_contig(&Vsx[lrow][ks * 32 + koff]);
    v16bf bf1 = frag_contig(&Vsx[16 + lrow][ks * 32 + koff]);
    oacc0 = WMMA_BF16(af, bf0, oacc0);
    oacc1 = WMMA_BF16(af, bf1, oacc1);
    __syncthreads();
  }
  #pragma unroll
  for (int r = 0; r < 8; ++r) {
    int jrow = j0 + r + hs;
    size_t ob = ((size_t)i * n + jrow) * 128 + h * 32;
    o[ob + lrow]      = oacc0[r];
    o[ob + 16 + lrow] = oacc1[r];
  }
}

// ======================= host orchestration ================================
// Input index map. Assumes harness passes setup_inputs() in insertion order
// (m, z, msa_mask, pair_mask, params) with params leaves flattened jax-style
// (dict keys sorted alphabetically, {b,w}/{b,g} within each leaf-dict).
enum {
  IN_M = 0, IN_Z = 1, IN_MSA_MASK = 2, IN_PAIR_MASK = 3,
  MSA_L1_B = 4, MSA_L1_W, MSA_L2_B, MSA_L2_W, MSA_LN_B, MSA_LN_G,
  OPM_LA_B = 10, OPM_LA_W, OPM_LB_B, OPM_LB_W, OPM_LN_B, OPM_LN_G, OPM_LO_B, OPM_LO_W,
  PT_L1_B = 18, PT_L1_W, PT_L2_B, PT_L2_W, PT_LN_B, PT_LN_G,
  TAE_BASE = 24,  // g.b,g.w,k.b,k.w,ln.b,ln.g,o.b,o.w,q.b,q.w,tb.b,tb.w,v.b,v.w
  TAS_BASE = 38,
  TMI_BASE = 52,  // ag.b,ag.w,ap.b,ap.w,bg.b,bg.w,bp.b,bp.w,g.b,g.w,ln_in.b,.g,ln_out.b,.g,p.b,p.w
  TMO_BASE = 68
};

extern "C" void kernel_launch(void* const* d_in, const int* in_sizes, int n_in,
                              void* d_out, int out_size, void* d_ws, size_t ws_size,
                              hipStream_t stream) {
  (void)in_sizes; (void)n_in; (void)out_size; (void)ws_size;
  const size_t NM = (size_t)N_SEQ_C * N_RES_C * CM_C;  // 8388608
  const size_t NZ = (size_t)N_RES_C * N_RES_C * CZ_C;  // 8388608
  float* m_out = (float*)d_out;
  float* z_out = m_out + NM;

  float* S0 = (float*)d_ws;
  float* S1 = S0 + NM;
  float* S2 = S1 + NM;
  float* S3 = S2 + NM;
  float* S4 = S3 + NM;
  float* S5 = S4 + NM;             // 33554432 floats (hidden / OPM-o-bf16 / attn-o)
  float* extra = S5 + 33554432;
  float* normbuf = extra; extra += 65536;
  float* tbbuf   = extra; extra += 262144;
  float* maskT   = extra; extra += 65536;
  __bf16* wp = (__bf16*)extra;     // bf16 weight arena (~2.4 MB used)

  const float* msa_mask  = (const float*)d_in[IN_MSA_MASK];
  const float* pair_mask = (const float*)d_in[IN_PAIR_MASK];
  auto inP = [&](int i) -> const float* { return (const float*)d_in[i]; };

  // Convert weight fp32[K,N] -> bf16[K,Npad], Npad rounded up to 64.
  auto conv = [&](int idx, int K, int N) -> const __bf16* {
    int Npad = (N + 63) & ~63;
    __bf16* dst = wp; wp += (size_t)K * Npad;
    int total = K * Npad;
    conv_pad_kernel<<<(total + 255) / 256, 256, 0, stream>>>(inP(idx), dst, K, N, Npad);
    return dst;
  };
  auto gemm = [&](const float* A, const __bf16* Abf, const __bf16* W, const float* bias,
                  float* Y, int M, int N, int K, int act) {
    int Npad = (N + 63) & ~63;
    dim3 g(Npad / 64, M / 128);
    if (Abf)
      gemm_bf16_kernel<1><<<g, 256, 0, stream>>>(A, Abf, W, bias, Y, M, N, Npad, K, act);
    else
      gemm_bf16_kernel<0><<<g, 256, 0, stream>>>(A, Abf, W, bias, Y, M, N, Npad, K, act);
  };
  auto lnorm = [&](const float* x, int gi, int bi, float* y, int rows, int C) {
    ln_kernel<<<rows, 256, 0, stream>>>(x, inP(gi), inP(bi), y, C);
  };
  auto mul3 = [&](float* y, const float* b, const float* tok, size_t total, int C) {
    ew_mul3_kernel<<<(unsigned)((total + 255) / 256), 256, 0, stream>>>(y, b, tok, total, C);
  };
  auto resadd = [&](float* z, const float* x, const float* tok, size_t total, int C) {
    ew_resadd_kernel<<<(unsigned)((total + 255) / 256), 256, 0, stream>>>(z, x, tok, total, C);
  };

  hipMemcpyAsync(m_out, d_in[IN_M], NM * sizeof(float), hipMemcpyDeviceToDevice, stream);
  hipMemcpyAsync(z_out, d_in[IN_Z], NZ * sizeof(float), hipMemcpyDeviceToDevice, stream);

  // ---- MSA transition: m += l2(relu(l1(LN(m)))) * mask ----
  lnorm(m_out, MSA_LN_G, MSA_LN_B, S0, N_SEQ_C * N_RES_C, CM_C);
  gemm(S0, nullptr, conv(MSA_L1_W, 256, 1024), inP(MSA_L1_B), S5, 32768, 1024, 256, 1);
  gemm(S5, nullptr, conv(MSA_L2_W, 1024, 256), inP(MSA_L2_B), S1, 32768, 256, 1024, 0);
  resadd(m_out, S1, msa_mask, NM, CM_C);

  // ---- Outer product mean ----
  lnorm(m_out, OPM_LN_G, OPM_LN_B, S0, N_SEQ_C * N_RES_C, CM_C);
  {
    float* abuf = S1;
    float* bbuf = S1 + (size_t)N_SEQ_C * N_RES_C * 32;
    gemm(S0, nullptr, conv(OPM_LA_W, 256, 32), inP(OPM_LA_B), abuf, 32768, 32, 256, 0);
    mul3(abuf, nullptr, msa_mask, (size_t)32768 * 32, 32);
    gemm(S0, nullptr, conv(OPM_LB_W, 256, 32), inP(OPM_LB_B), bbuf, 32768, 32, 256, 0);
    mul3(bbuf, nullptr, msa_mask, (size_t)32768 * 32, 32);
    __bf16* obf = (__bf16*)S5;
    opm_outer_kernel<<<dim3(N_RES_C, N_RES_C), 128, 0, stream>>>(abuf, bbuf, obf, N_SEQ_C, N_RES_C);
    gemm(nullptr, obf, conv(OPM_LO_W, 1024, 128), inP(OPM_LO_B), S2, 65536, 128, 1024, 0);
    norm_outer_kernel<<<256, 256, 0, stream>>>(msa_mask, normbuf, N_SEQ_C, N_RES_C);
    ew_opmfinal_kernel<<<(unsigned)((NZ + 255) / 256), 256, 0, stream>>>(z_out, S2, normbuf, NZ, CZ_C);
  }

  // ---- Triangle multiplications ----
  auto run_trimul = [&](int B, int outgoing) {
    lnorm(z_out, B + 11, B + 10, S0, 65536, CZ_C);                              // ln_in
    gemm(S0, nullptr, conv(B + 3, 128, 128), inP(B + 2), S1, 65536, 128, 128, 0);  // ap
    gemm(S0, nullptr, conv(B + 1, 128, 128), inP(B + 0), S2, 65536, 128, 128, 2);  // ag
    mul3(S1, S2, pair_mask, NZ, CZ_C);                                          // a
    gemm(S0, nullptr, conv(B + 7, 128, 128), inP(B + 6), S2, 65536, 128, 128, 0);  // bp
    gemm(S0, nullptr, conv(B + 5, 128, 128), inP(B + 4), S3, 65536, 128, 128, 2);  // bg
    mul3(S2, S3, pair_mask, NZ, CZ_C);                                          // b
    trimul_einsum_kernel<<<dim3(16, 16, 16), 256, 0, stream>>>(S1, S2, S3, N_RES_C, CZ_C, outgoing);
    lnorm(S3, B + 13, B + 12, S4, 65536, CZ_C);                                 // ln_out
    gemm(S0, nullptr, conv(B + 9, 128, 128), inP(B + 8), S3, 65536, 128, 128, 2);  // g
    gemm(S4, nullptr, conv(B + 15, 128, 128), inP(B + 14), S1, 65536, 128, 128, 0); // p
    ew_resmul_kernel<<<(unsigned)((NZ + 255) / 256), 256, 0, stream>>>(z_out, S3, S1, NZ);
  };
  run_trimul(TMO_BASE, 1);
  run_trimul(TMI_BASE, 0);

  // ---- Triangle attentions ----
  auto run_triatt = [&](int B, int starting) {
    const float* maskp = pair_mask;
    const float* zin = z_out;
    if (!starting) {
      transpose_nn_kernel<<<(unsigned)((NZ + 255) / 256), 256, 0, stream>>>(z_out, S1, N_RES_C, CZ_C);
      transpose_nn_kernel<<<256, 256, 0, stream>>>(pair_mask, maskT, N_RES_C, 1);
      zin = S1; maskp = maskT;
    }
    lnorm(zin, B + 5, B + 4, S0, 65536, CZ_C);
    gemm(S0, nullptr, conv(B + 9, 128, 128), inP(B + 8), S1, 65536, 128, 128, 0);   // q
    gemm(S0, nullptr, conv(B + 3, 128, 128), inP(B + 2), S2, 65536, 128, 128, 0);   // k
    gemm(S0, nullptr, conv(B + 13, 128, 128), inP(B + 12), S3, 65536, 128, 128, 0); // v
    gemm(S0, nullptr, conv(B + 1, 128, 128), inP(B + 0), S4, 65536, 128, 128, 2);   // g
    gemm(S0, nullptr, conv(B + 11, 128, 4), inP(B + 10), tbbuf, 65536, 4, 128, 0);  // tri bias
    tri_att_kernel<<<dim3(4, 4, N_RES_C), 128, 0, stream>>>(S1, S2, S3, tbbuf, maskp, S5, N_RES_C);
    mul3(S5, S4, nullptr, NZ, CZ_C);                                          // g * o
    gemm(S5, nullptr, conv(B + 7, 128, 128), inP(B + 6), S2, 65536, 128, 128, 0);   // out proj
    if (starting) resadd(z_out, S2, nullptr, NZ, CZ_C);
    else ew_addT_kernel<<<(unsigned)((NZ + 255) / 256), 256, 0, stream>>>(z_out, S2, N_RES_C, CZ_C);
  };
  run_triatt(TAS_BASE, 1);
  run_triatt(TAE_BASE, 0);

  // ---- Pair transition ----
  lnorm(z_out, PT_LN_G, PT_LN_B, S0, 65536, CZ_C);
  gemm(S0, nullptr, conv(PT_L1_W, 128, 512), inP(PT_L1_B), S5, 65536, 512, 128, 1);
  gemm(S5, nullptr, conv(PT_L2_W, 512, 128), inP(PT_L2_B), S1, 65536, 128, 512, 0);
  resadd(z_out, S1, pair_mask, NZ, CZ_C);
}
